// GIN_88888643158270
// MI455X (gfx1250) — matrix operand order repaired and checked by reading, not compiled, not run on hardware
//
#include <hip/hip_runtime.h>
#include <hip/hip_bf16.h>

typedef __attribute__((ext_vector_type(16))) __bf16 v16bf;
typedef __attribute__((ext_vector_type(8)))  float  v8f;

#define HDIM 128
#define KDIM 128
#define LSTR 132   // LDS row stride in bf16 elements (keeps dword alignment: 132*2=264, /4 ok)

__device__ __forceinline__ unsigned short f2bf(float f) {
    unsigned int u = __float_as_uint(f);
    u += 0x7fffu + ((u >> 16) & 1u);   // round-to-nearest-even
    return (unsigned short)(u >> 16);
}

// C[n x ncols] = A[n x 128] @ W[128 x ncols] + bias, optional relu.
// Block: 128 threads = 4 waves. Tile: 64 rows x (NT*16) cols, K=128 staged fully in LDS.
template <int NT>
__global__ __launch_bounds__(128) void gemm_bias(
    const float* __restrict__ A, const float* __restrict__ W,
    const float* __restrict__ bias, float* __restrict__ C,
    int n, int ncols, int relu)
{
    __shared__ unsigned short As[64 * LSTR];    // A tile, bf16, row-major [row][k]
    __shared__ unsigned short Ws[128 * LSTR];   // W transposed, bf16 [col][k]

    const int tid  = threadIdx.x;
    const int row0 = blockIdx.x * 64;

    // Stage A tile (64x128) -> LDS bf16
    for (int idx = tid; idx < 64 * KDIM; idx += 128) {
        int r = idx >> 7, k = idx & 127;
        float v = (row0 + r < n) ? A[(size_t)(row0 + r) * KDIM + k] : 0.0f;
        As[r * LSTR + k] = f2bf(v);
    }
    // Stage W (128 x ncols) -> LDS transposed bf16 [col][k], zero-pad cols >= ncols
    for (int idx = tid; idx < 128 * KDIM; idx += 128) {
        int k = idx >> 7, c = idx & 127;
        float v = (c < ncols) ? W[(size_t)k * ncols + c] : 0.0f;
        Ws[c * LSTR + k] = f2bf(v);
    }
    __syncthreads();

    const int wave = tid >> 5;        // 0..3 -> row strip
    const int lane = tid & 31;
    const int lhi  = lane >> 4;       // 0/1
    const int llo  = lane & 15;
    const int rowt = wave * 16;

    v8f acc[NT];
#pragma unroll
    for (int t = 0; t < NT; ++t) acc[t] = (v8f){0,0,0,0,0,0,0,0};

#pragma unroll
    for (int k0 = 0; k0 < KDIM; k0 += 32) {
        // A fragment: 16x32 bf16, ISA layout (lanes 0-15: M=0..15, K 0-7/16-23; lanes 16-31: K 8-15/24-31)
        union { unsigned int u[8]; v16bf v; } af;
        const unsigned int* Arow = (const unsigned int*)&As[(rowt + llo) * LSTR];
#pragma unroll
        for (int q = 0; q < 8; ++q) {
            int k = k0 + ((q >= 4) ? 16 : 0) + (lhi ? 8 : 0) + 2 * (q & 3);
            af.u[q] = Arow[k >> 1];
        }
#pragma unroll
        for (int t = 0; t < NT; ++t) {
            // B fragment: 32x16 bf16 (lanes 0-15: K 0-7/16-23, lanes 16-31: K 8-15/24-31, col = lane%16)
            union { unsigned int u[8]; v16bf v; } bfr;
            const unsigned int* Brow = (const unsigned int*)&Ws[(t * 16 + llo) * LSTR];
#pragma unroll
            for (int q = 0; q < 8; ++q) {
                int k = k0 + ((q >= 4) ? 16 : 0) + (lhi ? 8 : 0) + 2 * (q & 3);
                bfr.u[q] = Brow[k >> 1];
            }
            acc[t] = __builtin_amdgcn_wmma_f32_16x16x32_bf16(
                false, af.v, false, bfr.v, (short)0, acc[t], false, false);
        }
    }

    // Store: C/D layout — lanes 0-15: N=lane, M = vgpr + (lhi? +8)
#pragma unroll
    for (int t = 0; t < NT; ++t) {
        int col = t * 16 + llo;
        if (col >= ncols) continue;
        float b = bias[col];
#pragma unroll
        for (int r = 0; r < 8; ++r) {
            int row = row0 + rowt + r + lhi * 8;
            if (row < n) {
                float o = acc[t][r] + b;
                if (relu) o = fmaxf(o, 0.0f);
                C[(size_t)row * ncols + col] = o;
            }
        }
    }
}

// Per-column sum & sum-of-squares (atomics into stats[0..127], stats[128..255])
__global__ __launch_bounds__(256) void col_stats(
    const float* __restrict__ X, int n, float* __restrict__ stats)
{
    int c    = threadIdx.x & 127;
    int half = threadIdx.x >> 7;
    int row0 = blockIdx.x * 512 + half;
    float s = 0.f, s2 = 0.f;
    for (int i = 0; i < 256; ++i) {
        int r = row0 + 2 * i;
        if (r < n) {
            float v = X[(size_t)r * HDIM + c];
            s += v; s2 += v * v;
        }
    }
    atomicAdd(&stats[c], s);
    atomicAdd(&stats[128 + c], s2);
}

// stats -> (scale, shift) in place
__global__ void bn_finalize(float* __restrict__ stats,
                            const float* __restrict__ g,
                            const float* __restrict__ be, int n)
{
    int c = threadIdx.x;   // 128 threads
    float inv_n = 1.0f / (float)n;
    float mean = stats[c] * inv_n;
    float var  = stats[128 + c] * inv_n - mean * mean;
    float sc   = g[c] * rsqrtf(var + 1e-5f);
    stats[c]        = sc;
    stats[128 + c]  = be[c] - mean * sc;
}

__global__ __launch_bounds__(256) void bn_relu_apply(
    const float* __restrict__ X, const float* __restrict__ stats,
    float* __restrict__ Y, long long total)
{
    long long i = (long long)blockIdx.x * 256 + threadIdx.x;
    if (i >= total) return;
    int c = (int)(i & 127);
    float v = X[i] * stats[c] + stats[128 + c];
    Y[i] = fmaxf(v, 0.0f);
}

// One wave per directed edge; lane handles 4 contiguous features.
__global__ __launch_bounds__(256) void edge_scatter(
    const float* __restrict__ Hh, const long long* __restrict__ ei,
    long long e, float* __restrict__ Agg)
{
    long long t = (long long)blockIdx.x * 256 + threadIdx.x;
    int lane = (int)(t & 31);
    long long w = t >> 5;
    if (w >= 2 * e) return;
    long long s, d;
    if (w < e) { s = ei[w];     d = ei[e + w]; }
    else       { long long i = w - e; s = ei[e + i]; d = ei[i]; }
    const float4 v = *(const float4*)(Hh + s * HDIM + lane * 4);
    float* out = Agg + d * HDIM + lane * 4;
    atomicAdd(out + 0, v.x);
    atomicAdd(out + 1, v.y);
    atomicAdd(out + 2, v.z);
    atomicAdd(out + 3, v.w);
}

// z = (2 + eps) * h + segsum   (agg(h) = segsum + h, plus (1+eps)*h)
__global__ __launch_bounds__(256) void combine_eps(
    const float* __restrict__ Hh, const float* __restrict__ Agg,
    const float* __restrict__ eps, float* __restrict__ Z, long long total)
{
    long long i = (long long)blockIdx.x * 256 + threadIdx.x;
    if (i >= total) return;
    float a = 2.0f + eps[0];
    Z[i] = a * Hh[i] + Agg[i];
}

extern "C" void kernel_launch(void* const* d_in, const int* in_sizes, int n_in,
                              void* d_out, int out_size, void* d_ws, size_t ws_size,
                              hipStream_t stream)
{
    const float*     x    = (const float*)d_in[0];
    const long long* ei   = (const long long*)d_in[1];
    const float*     eps1 = (const float*)d_in[2];
    const float*     eps2 = (const float*)d_in[3];
    const float *W_in = (const float*)d_in[4],  *b_in = (const float*)d_in[5];
    const float *g_in = (const float*)d_in[6],  *be_in= (const float*)d_in[7];
    const float *W1a  = (const float*)d_in[8],  *b1a  = (const float*)d_in[9];
    const float *W1b  = (const float*)d_in[10], *b1b  = (const float*)d_in[11];
    const float *g1   = (const float*)d_in[12], *be1  = (const float*)d_in[13];
    const float *W2a  = (const float*)d_in[14], *b2a  = (const float*)d_in[15];
    const float *W2b  = (const float*)d_in[16], *b2b  = (const float*)d_in[17];
    const float *g2   = (const float*)d_in[18], *be2  = (const float*)d_in[19];
    const float *Wh   = (const float*)d_in[20], *bh   = (const float*)d_in[21];

    const int n = in_sizes[0] / HDIM;                 // 100000
    const long long e = (long long)(in_sizes[1] / 2); // 500000
    const long long total = (long long)n * HDIM;

    float* buf0  = (float*)d_ws;          // activations
    float* buf1  = buf0 + total;          // gemm out / z
    float* buf2  = buf1 + total;          // aggregation accumulator
    float* stats = buf2 + total;          // 256 floats

    const int gemm_blocks = (n + 63) / 64;
    const int ew_blocks   = (int)((2 * e * 32 + 255) / 256);
    const int el_blocks   = (int)((total + 255) / 256);
    const int cs_blocks   = (n + 511) / 512;

    auto bn_stage = [&](const float* X, const float* g, const float* be, float* Y) {
        hipMemsetAsync(stats, 0, 256 * sizeof(float), stream);
        col_stats<<<cs_blocks, 256, 0, stream>>>(X, n, stats);
        bn_finalize<<<1, 128, 0, stream>>>(stats, g, be, n);
        bn_relu_apply<<<el_blocks, 256, 0, stream>>>(X, stats, Y, total);
    };
    auto agg_stage = [&](const float* Hh, const float* eps, float* Z) {
        hipMemsetAsync(buf2, 0, total * sizeof(float), stream);
        edge_scatter<<<ew_blocks, 256, 0, stream>>>(Hh, ei, e, buf2);
        combine_eps<<<el_blocks, 256, 0, stream>>>(Hh, buf2, eps, Z, total);
    };

    // Layer 0: x0 = relu(BN(x @ W_in + b_in))
    gemm_bias<8><<<gemm_blocks, 128, 0, stream>>>(x, W_in, b_in, buf1, n, HDIM, 0);
    bn_stage(buf1, g_in, be_in, buf0);                       // buf0 = x0

    // Layer 1
    agg_stage(buf0, eps1, buf1);                             // buf1 = z1
    gemm_bias<8><<<gemm_blocks, 128, 0, stream>>>(buf1, W1a, b1a, buf0, n, HDIM, 1);
    gemm_bias<8><<<gemm_blocks, 128, 0, stream>>>(buf0, W1b, b1b, buf1, n, HDIM, 0);
    bn_stage(buf1, g1, be1, buf0);                           // buf0 = h1

    // Layer 2
    agg_stage(buf0, eps2, buf1);                             // buf1 = z2
    gemm_bias<8><<<gemm_blocks, 128, 0, stream>>>(buf1, W2a, b2a, buf0, n, HDIM, 1);
    gemm_bias<8><<<gemm_blocks, 128, 0, stream>>>(buf0, W2b, b2b, buf1, n, HDIM, 0);
    bn_stage(buf1, g2, be2, buf0);                           // buf0 = h2

    // Head: out = h2 @ Wh + bh   (ncols = 40 -> 3 column tiles)
    gemm_bias<3><<<gemm_blocks, 128, 0, stream>>>(buf0, Wh, bh, (float*)d_out, n, 40, 0);
}
